// GCN_51616916963749
// MI455X (gfx1250) — compile-verified, matrix operand order checked by reading, not run on hardware
//
#include <hip/hip_runtime.h>
#include <hip/hip_bf16.h>
#include <math.h>

typedef __attribute__((ext_vector_type(16))) __bf16 v16bf;
typedef __attribute__((ext_vector_type(8)))  float  v8f;
typedef __attribute__((ext_vector_type(4)))  float  v4f;
typedef v4f v4f_u __attribute__((aligned(4)));            // 4B-aligned 128-bit load
typedef __attribute__((ext_vector_type(4)))  unsigned u4;

__device__ __forceinline__ unsigned rnd_bf16(float f) {   // RNE, result in bits[31:16]
    unsigned u = __builtin_bit_cast(unsigned, f);
    return u + 0x7FFFu + ((u >> 16) & 1u);
}
__device__ __forceinline__ unsigned pack2_bf16(float lo, float hi) {
    return (rnd_bf16(lo) >> 16) | (rnd_bf16(hi) & 0xFFFF0000u);
}
__device__ __forceinline__ __bf16 cvt_bf16(float f) {
    unsigned short h = (unsigned short)(rnd_bf16(f) >> 16);
    return __builtin_bit_cast(__bf16, h);
}

// ---------------- degree / norm precompute ----------------
__global__ void k_deg_init(float* deg, int n) {
    int i = blockIdx.x * blockDim.x + threadIdx.x;
    if (i < n) deg[i] = 1.0f;                   // self-loop contribution
}
__global__ void k_deg_edges(const int* __restrict__ col, float* deg, int e) {
    int i = blockIdx.x * blockDim.x + threadIdx.x;
    if (i < e) atomicAdd(&deg[col[i]], 1.0f);
}
__global__ void k_dinv(float* deg, int n) {
    int i = blockIdx.x * blockDim.x + threadIdx.x;
    if (i < n) deg[i] = rsqrtf(deg[i]);         // deg >= 1 always
}

// ---------------- bf16 WMMA GEMM: C[M x Nreal] = A[M x K] @ B[K x Nreal] ----------------
#define BM 128
#define BN 64
#define BK 32
#define LDA 40
#define LDB 40

__global__ __launch_bounds__(256) void k_gemm_bf16(
    const float* __restrict__ A, const float* __restrict__ B, float* __restrict__ C,
    int M, int K, int Nreal, int ldc)
{
    __shared__ __bf16 Al[BM][LDA];
    __shared__ __bf16 Bl[BN][LDB];      // stored transposed: Bl[n][k]

    const int tid  = threadIdx.x;
    const int lane = tid & 31;
    const int wave = tid >> 5;
    const int wm   = wave & 3;          // 4 row-groups of 32
    const int wn   = wave >> 2;         // 2 col-groups of 32
    const int m0   = blockIdx.x * BM;
    const int n0   = blockIdx.y * BN;

    v8f acc[2][2] = {};

    const int arow = tid >> 1;              // 0..127
    const int akh  = (tid & 1) * 16;        // 0 or 16
    const int bn_  = tid & 63;              // 0..63
    const int bk0  = (tid >> 6) * 8;        // 0,8,16,24

    const bool full_m = (m0 + BM <= M);
    const bool full_n = (n0 + BN <= Nreal);

    for (int k0 = 0; k0 < K; k0 += BK) {
        const bool full_k = (k0 + BK <= K);

        // ---- A tile: 128x32 f32 -> bf16 (each thread: 16 contiguous floats) ----
        if (full_m & full_k) {
            const float* ap = A + (size_t)(m0 + arow) * K + k0 + akh;
            v4f f0 = *(const v4f_u*)(ap + 0);
            v4f f1 = *(const v4f_u*)(ap + 4);
            v4f f2 = *(const v4f_u*)(ap + 8);
            v4f f3 = *(const v4f_u*)(ap + 12);
            u4 p0 = { pack2_bf16(f0.x, f0.y), pack2_bf16(f0.z, f0.w),
                      pack2_bf16(f1.x, f1.y), pack2_bf16(f1.z, f1.w) };
            u4 p1 = { pack2_bf16(f2.x, f2.y), pack2_bf16(f2.z, f2.w),
                      pack2_bf16(f3.x, f3.y), pack2_bf16(f3.z, f3.w) };
            u4* dst = (u4*)&Al[arow][akh];   // 16B aligned (80B row stride)
            dst[0] = p0;
            dst[1] = p1;
        } else {
            int gr = m0 + arow;
#pragma unroll
            for (int j = 0; j < 16; ++j) {
                int gk = k0 + akh + j;
                float v = (gr < M && gk < K) ? A[(size_t)gr * K + gk] : 0.0f;
                Al[arow][akh + j] = cvt_bf16(v);
            }
        }
        // ---- B tile: 32x64 f32, transposed into LDS (each thread: 8 k-strided) ----
        if (full_n & full_k) {
            const float* bp = B + (size_t)(k0 + bk0) * Nreal + n0 + bn_;
            float t[8];
#pragma unroll
            for (int j = 0; j < 8; ++j) t[j] = bp[(size_t)j * Nreal];
            u4 pb = { pack2_bf16(t[0], t[1]), pack2_bf16(t[2], t[3]),
                      pack2_bf16(t[4], t[5]), pack2_bf16(t[6], t[7]) };
            *(u4*)&Bl[bn_][bk0] = pb;       // 16B aligned
        } else {
            int gn = n0 + bn_;
#pragma unroll
            for (int j = 0; j < 8; ++j) {
                int gk = k0 + bk0 + j;
                float v = (gn < Nreal && gk < K) ? B[(size_t)gk * Nreal + gn] : 0.0f;
                Bl[bn_][bk0 + j] = cvt_bf16(v);
            }
        }
        __syncthreads();

        const int khalf = (lane >= 16) ? 8 : 0;   // ISA 16-bit A/B lane K layout
        const int rl    = lane & 15;

        v16bf bfrag[2];
#pragma unroll
        for (int tj = 0; tj < 2; ++tj) {
            int c = wn * 32 + tj * 16 + rl;
#pragma unroll
            for (int j = 0; j < 8; ++j) bfrag[tj][j]     = Bl[c][khalf + j];
#pragma unroll
            for (int j = 0; j < 8; ++j) bfrag[tj][8 + j] = Bl[c][khalf + 16 + j];
        }
#pragma unroll
        for (int ti = 0; ti < 2; ++ti) {
            v16bf a;
            int r = wm * 32 + ti * 16 + rl;
#pragma unroll
            for (int j = 0; j < 8; ++j) a[j]     = Al[r][khalf + j];
#pragma unroll
            for (int j = 0; j < 8; ++j) a[8 + j] = Al[r][khalf + 16 + j];
#pragma unroll
            for (int tj = 0; tj < 2; ++tj) {
                acc[ti][tj] = __builtin_amdgcn_wmma_f32_16x16x32_bf16(
                    false, a, false, bfrag[tj], (short)0, acc[ti][tj], false, false);
            }
        }
        __syncthreads();
    }

    // store: lane n = lane&15, M half = (lane>>4)*8, VGPR v -> row m = half+v
    const int nlane = lane & 15;
    const int mhalf = (lane >> 4) * 8;
#pragma unroll
    for (int ti = 0; ti < 2; ++ti)
#pragma unroll
        for (int tj = 0; tj < 2; ++tj) {
            int col = n0 + wn * 32 + tj * 16 + nlane;
            if (col >= Nreal) continue;
#pragma unroll
            for (int v = 0; v < 8; ++v) {
                int row = m0 + wm * 32 + ti * 16 + mhalf + v;
                if (row < M) C[(size_t)row * ldc + col] = acc[ti][tj][v];
            }
        }
}

// ---------------- agg init: self-loop + bias ----------------
__global__ void k_agg_init(const float* __restrict__ Hlin, const float* __restrict__ dinv,
                           const float* __restrict__ bias, float* __restrict__ agg,
                           int n, int F, int ld)
{
    long long idx = (long long)blockIdx.x * blockDim.x + threadIdx.x;
    if (idx >= (long long)n * F) return;
    int i = (int)(idx / F), f = (int)(idx % F);
    float d = dinv[i];
    agg[(size_t)i * ld + f] = Hlin[(size_t)i * ld + f] * d * d + bias[f];
}

// ---------------- edge scatter: agg[col] += Hlin[row] * dinv[row]*dinv[col] ----------------
template <int VEC>
__global__ void k_scatter(const int* __restrict__ row, const int* __restrict__ col,
                          const float* __restrict__ dinv,
                          const float* __restrict__ Hlin, float* __restrict__ agg,
                          int E, int F, int ld)
{
    int chunks = F / VEC;
    long long idx = (long long)blockIdx.x * blockDim.x + threadIdx.x;
    if (idx >= (long long)E * chunks) return;
    int e = (int)(idx / chunks);
    int c = (int)(idx % chunks);
    int r = row[e], t = col[e];
    float nrm = dinv[r] * dinv[t];
    const float* src = Hlin + (size_t)r * ld + c * VEC;
    float*       dst = agg  + (size_t)t * ld + c * VEC;
#pragma unroll
    for (int j = 0; j < VEC; ++j) atomicAdd(dst + j, src[j] * nrm);
}

__global__ void k_relu(float* x, long long n) {
    long long i = (long long)blockIdx.x * blockDim.x + threadIdx.x;
    if (i < n) x[i] = fmaxf(x[i], 0.0f);
}

__global__ void k_logsoftmax(const float* __restrict__ agg, float* __restrict__ out,
                             int n, int C, int ld)
{
    int i = blockIdx.x * blockDim.x + threadIdx.x;
    if (i >= n) return;
    float m = -INFINITY;
    for (int c = 0; c < C; ++c) m = fmaxf(m, agg[(size_t)i * ld + c]);
    float s = 0.0f;
    for (int c = 0; c < C; ++c) s += expf(agg[(size_t)i * ld + c] - m);
    float ls = logf(s);
    for (int c = 0; c < C; ++c) out[(size_t)i * C + c] = agg[(size_t)i * ld + c] - m - ls;
}

// ---------------- host launcher ----------------
extern "C" void kernel_launch(void* const* d_in, const int* in_sizes, int n_in,
                              void* d_out, int out_size, void* d_ws, size_t ws_size,
                              hipStream_t stream)
{
    const float* x  = (const float*)d_in[0];
    const int*   ei = (const int*)  d_in[1];
    const float* W1 = (const float*)d_in[2];
    const float* b1 = (const float*)d_in[3];
    const float* W2 = (const float*)d_in[4];
    const float* b2 = (const float*)d_in[5];
    const float* W3 = (const float*)d_in[6];
    const float* b3 = (const float*)d_in[7];
    float* out = (float*)d_out;

    const int H   = in_sizes[3];            // 256
    const int C   = in_sizes[7];            // 7
    const int FIN = in_sizes[2] / H;        // 1433
    const int N   = in_sizes[0] / FIN;      // 50000
    const int E   = in_sizes[1] / 2;        // 800000

    const int* row = ei;
    const int* col = ei + E;

    const int LD3 = 16;                     // padded layer-3 leading dim

    // carve workspace
    char* p = (char*)d_ws;
    auto carve = [&](size_t bytes) { void* r = (void*)p; p += (bytes + 255) & ~(size_t)255; return r; };
    float* dinv = (float*)carve(sizeof(float) * (size_t)N);
    float* buf0 = (float*)carve(sizeof(float) * (size_t)N * H);   // GEMM out (Hlin)
    float* buf1 = (float*)carve(sizeof(float) * (size_t)N * H);   // agg / activation
    float* buf2 = (float*)carve(sizeof(float) * (size_t)N * LD3); // layer-3 Hlin
    float* buf3 = (float*)carve(sizeof(float) * (size_t)N * LD3); // layer-3 agg

    const int TB = 256;
    dim3 blk(TB);

    // ---- gcn_norm ----
    k_deg_init <<<dim3((N + TB - 1) / TB), blk, 0, stream>>>(dinv, N);
    k_deg_edges<<<dim3((E + TB - 1) / TB), blk, 0, stream>>>(col, dinv, E);
    k_dinv     <<<dim3((N + TB - 1) / TB), blk, 0, stream>>>(dinv, N);

    const long long nH  = (long long)N * H;
    const int gNH = (int)((nH + TB - 1) / TB);
    const long long sc1 = (long long)E * (H / 4);
    const int gSC = (int)((sc1 + TB - 1) / TB);

    dim3 g1((N + BM - 1) / BM, (H + BN - 1) / BN);

    // ---- layer 1 ----
    k_gemm_bf16<<<g1, blk, 0, stream>>>(x, W1, buf0, N, FIN, H, H);
    k_agg_init <<<dim3(gNH), blk, 0, stream>>>(buf0, dinv, b1, buf1, N, H, H);
    k_scatter<4><<<dim3(gSC), blk, 0, stream>>>(row, col, dinv, buf0, buf1, E, H, H);
    k_relu     <<<dim3(gNH), blk, 0, stream>>>(buf1, nH);

    // ---- layer 2 ----
    k_gemm_bf16<<<g1, blk, 0, stream>>>(buf1, W2, buf0, N, H, H, H);
    k_agg_init <<<dim3(gNH), blk, 0, stream>>>(buf0, dinv, b2, buf1, N, H, H);
    k_scatter<4><<<dim3(gSC), blk, 0, stream>>>(row, col, dinv, buf0, buf1, E, H, H);
    k_relu     <<<dim3(gNH), blk, 0, stream>>>(buf1, nH);

    // ---- layer 3 (N=7 padded into BN=64 tile, ldc=16) ----
    dim3 g3((N + BM - 1) / BM, 1);
    k_gemm_bf16<<<g3, blk, 0, stream>>>(buf1, W3, buf2, N, H, C, LD3);
    const long long nC = (long long)N * C;
    k_agg_init <<<dim3((int)((nC + TB - 1) / TB)), blk, 0, stream>>>(buf2, dinv, b3, buf3, N, C, LD3);
    const long long sc3 = (long long)E * C;
    k_scatter<1><<<dim3((int)((sc3 + TB - 1) / TB)), blk, 0, stream>>>(row, col, dinv, buf2, buf3, E, C, LD3);
    k_logsoftmax<<<dim3((N + TB - 1) / TB), blk, 0, stream>>>(buf3, out, N, C, LD3);
}
